// LSTM_Encoder_51410758533279
// MI455X (gfx1250) — compile-verified, matrix operand order checked by reading, not run on hardware
//
#include <hip/hip_runtime.h>
#include <hip/hip_bf16.h>

// ---------------- CDNA5 types ----------------
typedef __bf16 bf16;
typedef __bf16 v16bf __attribute__((ext_vector_type(16)));
typedef __bf16 v8bf  __attribute__((ext_vector_type(8)));
typedef float  v8f   __attribute__((ext_vector_type(8)));

// Problem constants
#define TT   256
#define BB   64
#define EE   512
#define HH   1024
#define G4H  4096
#define ZZ   128
#define NROW (TT * BB)          // 16384 rows of hidden
#define HIDK 4096               // 2*L*H

__device__ __forceinline__ float sigf(float x) { return 1.0f / (1.0f + __expf(-x)); }

// Opaque copy of a loop bound: stops full unroll of the K loop, which in turn
// stops LICM from hoisting the (t-invariant) weight-fragment loads out of the
// persistent t-loop -- that hoisting is what caused the mega-spill to scratch.
__device__ __forceinline__ int opaque(int v) {
    asm volatile("" : "+s"(v));
    return v;
}

// ---- WMMA bf16 fragment load from a lane-adjusted pointer ------------------
// 16-bit A (16x32) per-lane layout (ISA 7.12.2): lane<16 holds K=[k0..k0+7,
// k0+16..k0+23] of row (lane&15); lane>=16 holds K offsets +8. B (32x16) is
// the mirrored layout with N=(lane&15). Since gates = h * W^T with row-major
// W[4H,K], BOTH operands read contiguous K -> two b128 loads per lane.
__device__ __forceinline__ v16bf load_frag_ptr(const bf16* __restrict__ p) {
    v8bf lo = *(const v8bf*)(p);
    v8bf hi = *(const v8bf*)(p + 16);
    v16bf f;
#pragma unroll
    for (int i = 0; i < 8; ++i) { f[i] = lo[i]; f[i + 8] = hi[i]; }
    return f;
}

// Lane-adjusted pointer for a fragment whose rows start at row0 (ld = K).
__device__ __forceinline__ const bf16* frag_base(const bf16* __restrict__ base,
                                                 int row0, int ld) {
    const int lane = threadIdx.x & 31;
    return base + (size_t)(row0 + (lane & 15)) * ld + ((lane >> 4) << 3);
}

__device__ __forceinline__ v8f wmma_bf16(v16bf a, v16bf b, v8f c) {
    // (neg_a, A, neg_b, B, c_mod, C, reuse_a, reuse_b)
    return __builtin_amdgcn_wmma_f32_16x16x32_bf16(false, a, false, b,
                                                   (short)0, c, false, false);
}

// Accumulate all four gate tiles (i,f,g,o) for one 16x16 (m,j) block.
// Rolled K loop (opaque bound), loop-carried pointers, max 9 live 8-VGPR
// tuples (4 acc + 5 fragments) -> spill-free; all 10 b128 loads of an
// iteration issue under s_clause before the 4 WMMAs consume them.
__device__ __forceinline__ void gemm_accum(v8f& acc0, v8f& acc1, v8f& acc2, v8f& acc3,
                                           const bf16* __restrict__ A, int mbase, int K,
                                           const bf16* __restrict__ W, int jbase) {
    const bf16* pa  = frag_base(A, mbase, K);
    const bf16* pb0 = frag_base(W, 0 * HH + jbase, K);
    const bf16* pb1 = frag_base(W, 1 * HH + jbase, K);
    const bf16* pb2 = frag_base(W, 2 * HH + jbase, K);
    const bf16* pb3 = frag_base(W, 3 * HH + jbase, K);

    const int kk = opaque(K);
    for (int k = 0; k < kk; k += 32) {
        v16bf a  = load_frag_ptr(pa);
        v16bf b0 = load_frag_ptr(pb0);
        v16bf b1 = load_frag_ptr(pb1);
        v16bf b2 = load_frag_ptr(pb2);
        v16bf b3 = load_frag_ptr(pb3);
        pa += 32; pb0 += 32; pb1 += 32; pb2 += 32; pb3 += 32;
        acc0 = wmma_bf16(a, b0, acc0);
        acc1 = wmma_bf16(a, b1, acc1);
        acc2 = wmma_bf16(a, b2, acc2);
        acc3 = wmma_bf16(a, b3, acc3);
    }
}

// In-register LSTM cell update for the wave-owned 16x16 block. Writes fp32 c
// state, bf16 h state (next-step WMMA A operand) and the bug-compatible
// hidden row: b' = 32*layer + m/2, k = (m&1)*2048 + j (h) / +1024 (c).
__device__ __forceinline__ void cell_update(v8f acc0, v8f acc1, v8f acc2, v8f acc3,
                                            float* __restrict__ c,
                                            bf16* __restrict__ hw,
                                            bf16* __restrict__ hidden,
                                            int t, int layer, int mbase, int jbase) {
    const int lane = threadIdx.x & 31;
    const int j    = jbase + (lane & 15);
    const int mo   = (lane >> 4) << 3;
#pragma unroll
    for (int r = 0; r < 8; ++r) {
        const int m   = mbase + mo + r;
        const int idx = m * HH + j;
        float cn = sigf(acc1[r]) * c[idx] + sigf(acc0[r]) * tanhf(acc2[r]);
        float h  = sigf(acc3[r]) * tanhf(cn);
        c[idx]  = cn;
        hw[idx] = (bf16)h;
        const int    bp = layer * 32 + (m >> 1);
        const int    kk = ((m & 1) << 11) + j;
        const size_t hb = ((size_t)t * BB + bp) * (size_t)HIDK;
        hidden[hb + kk]        = (bf16)h;
        hidden[hb + kk + 1024] = (bf16)cn;
    }
}

// Sense-reversing grid barrier (device-scope atomics); bar[0]=count bar[1]=gen.
__device__ __forceinline__ void grid_sync(unsigned* bar, unsigned nwg) {
    __syncthreads();
    if (threadIdx.x == 0) {
        unsigned gen = __hip_atomic_load(&bar[1], __ATOMIC_RELAXED, __HIP_MEMORY_SCOPE_AGENT);
        __threadfence();
        unsigned arrived = __hip_atomic_fetch_add(&bar[0], 1u, __ATOMIC_ACQ_REL,
                                                  __HIP_MEMORY_SCOPE_AGENT);
        if (arrived == nwg - 1) {
            __hip_atomic_store(&bar[0], 0u, __ATOMIC_RELAXED, __HIP_MEMORY_SCOPE_AGENT);
            __hip_atomic_fetch_add(&bar[1], 1u, __ATOMIC_RELEASE, __HIP_MEMORY_SCOPE_AGENT);
        } else {
            while (__hip_atomic_load(&bar[1], __ATOMIC_ACQUIRE,
                                     __HIP_MEMORY_SCOPE_AGENT) == gen)
                __builtin_amdgcn_s_sleep(1);
        }
    }
    __syncthreads();
}

// ---------------- small prep kernels ----------------
__global__ void zero_kernel(unsigned* p, int n) {
    int i = blockIdx.x * blockDim.x + threadIdx.x;
    if (i < n) p[i] = 0u;
}
__global__ void cvt_bf16_kernel(const float* __restrict__ s, bf16* __restrict__ d, int n) {
    int i = blockIdx.x * blockDim.x + threadIdx.x;
    if (i < n) d[i] = (bf16)s[i];
}
__global__ void bias_sum_kernel(const float* a, const float* b, float* o, int n) {
    int i = blockIdx.x * blockDim.x + threadIdx.x;
    if (i < n) o[i] = a[i] + b[i];
}
__global__ void embed_kernel(const int* __restrict__ tokens, const float* __restrict__ emb,
                             bf16* __restrict__ xbf) {
    int gid = blockIdx.x * blockDim.x + threadIdx.x;   // 64 threads per row, 8 elems each
    int row = gid >> 6;
    int col = (gid & 63) << 3;
    if (row >= NROW) return;
    const float* src = emb + (size_t)tokens[row] * EE + col;
    bf16* dst        = xbf + (size_t)row * EE + col;
#pragma unroll
    for (int i = 0; i < 8; ++i) dst[i] = (bf16)src[i];
}

// ---------------- persistent recurrence ----------------
// 32 WGs x 256 threads = 256 waves = 4 (M) x 64 (j) tiles; each wave owns one
// 16x16 (m,j) block and all four gates for it. h buffers ping-pong by t parity
// so the only cross-wave sync is 2 grid barriers per timestep. Weights
// (~30 MB bf16) stay resident in the 192 MB L2.
__global__ void lstm_persistent(const bf16* __restrict__ xbf,
                                const bf16* __restrict__ Wih0, const bf16* __restrict__ Whh0,
                                const bf16* __restrict__ Wih1, const bf16* __restrict__ Whh1,
                                const float* __restrict__ bias0, const float* __restrict__ bias1,
                                float* __restrict__ c0, float* __restrict__ c1,
                                bf16* __restrict__ h0buf, bf16* __restrict__ h1buf,
                                bf16* __restrict__ hidden, unsigned* bar) {
    const int lane  = threadIdx.x & 31;
    const int job   = blockIdx.x * 8 + (threadIdx.x >> 5);   // 0..255
    const int mbase = (job & 3) * 16;
    const int jbase = (job >> 2) * 16;
    const unsigned nwg = gridDim.x;
    const int HC = BB * HH;                                   // 65536 elems per h buffer

    for (int t = 0; t < TT; ++t) {
        const int p = t & 1;
        // ---- layer 0: gates = x_t*Wih0^T + h0*Whh0^T + bias0 ----
        {
            const bf16* h0r = h0buf + p * HC;
            bf16*       h0w = h0buf + (p ^ 1) * HC;
            const bf16* x   = xbf + (size_t)t * BB * EE;
            v8f a0, a1, a2, a3;
            {
                float bi  = bias0[0 * HH + jbase + (lane & 15)];
                float bf_ = bias0[1 * HH + jbase + (lane & 15)];
                float bg  = bias0[2 * HH + jbase + (lane & 15)];
                float bo  = bias0[3 * HH + jbase + (lane & 15)];
#pragma unroll
                for (int r = 0; r < 8; ++r) { a0[r] = bi; a1[r] = bf_; a2[r] = bg; a3[r] = bo; }
            }
            gemm_accum(a0, a1, a2, a3, x,   mbase, EE, Wih0, jbase);
            gemm_accum(a0, a1, a2, a3, h0r, mbase, HH, Whh0, jbase);
            cell_update(a0, a1, a2, a3, c0, h0w, hidden, t, 0, mbase, jbase);
        }
        grid_sync(bar, nwg);
        // ---- layer 1: gates = h0_new*Wih1^T + h1*Whh1^T + bias1 ----
        {
            const bf16* h0n = h0buf + (p ^ 1) * HC;           // just produced
            const bf16* h1r = h1buf + p * HC;
            bf16*       h1w = h1buf + (p ^ 1) * HC;
            v8f a0, a1, a2, a3;
            {
                float bi  = bias1[0 * HH + jbase + (lane & 15)];
                float bf_ = bias1[1 * HH + jbase + (lane & 15)];
                float bg  = bias1[2 * HH + jbase + (lane & 15)];
                float bo  = bias1[3 * HH + jbase + (lane & 15)];
#pragma unroll
                for (int r = 0; r < 8; ++r) { a0[r] = bi; a1[r] = bf_; a2[r] = bg; a3[r] = bo; }
            }
            gemm_accum(a0, a1, a2, a3, h0n, mbase, HH, Wih1, jbase);
            gemm_accum(a0, a1, a2, a3, h1r, mbase, HH, Whh1, jbase);
            cell_update(a0, a1, a2, a3, c1, h1w, hidden, t, 1, mbase, jbase);
        }
        grid_sync(bar, nwg);
    }
}

// ---------------- output projections ----------------
// [16384,4096] x [4096,128]^T for mean and log_std. 16384 wave-jobs, 1 tile
// (16x16) each, K-loop of 128 WMMAs (rolled; occupancy hides latency).
__global__ void proj_kernel(const bf16* __restrict__ hidden,
                            const bf16* __restrict__ Wm, const bf16* __restrict__ Wl,
                            const float* __restrict__ bm, const float* __restrict__ bl,
                            float* __restrict__ out) {
    const int lane = threadIdx.x & 31;
    const int wid  = blockIdx.x * 8 + (threadIdx.x >> 5);     // 0..16383
    const int which = wid >> 13;                              // 0=mean 1=log_std
    const int idx   = wid & 8191;
    const int mbase = (idx >> 3) * 16;
    const int nbase = (idx & 7) * 16;
    const bf16*  W    = which ? Wl : Wm;
    const float* bias = which ? bl : bm;
    float* ob = out + (size_t)which * ((size_t)NROW * ZZ);

    v8f acc;
    {
        float bn = bias[nbase + (lane & 15)];
#pragma unroll
        for (int r = 0; r < 8; ++r) acc[r] = bn;
    }
    const bf16* pa = frag_base(hidden, mbase, HIDK);
    const bf16* pb = frag_base(W,      nbase, HIDK);
    const int kk = opaque(HIDK);
    for (int k = 0; k < kk; k += 32) {
        v16bf a = load_frag_ptr(pa);
        v16bf b = load_frag_ptr(pb);
        pa += 32; pb += 32;
        acc = wmma_bf16(a, b, acc);
    }

    const int n  = nbase + (lane & 15);
    const int mo = (lane >> 4) << 3;
#pragma unroll
    for (int r = 0; r < 8; ++r)
        ob[(size_t)(mbase + mo + r) * ZZ + n] = acc[r];
}

// ---------------- host launcher ----------------
extern "C" void kernel_launch(void* const* d_in, const int* in_sizes, int n_in,
                              void* d_out, int out_size, void* d_ws, size_t ws_size,
                              hipStream_t stream) {
    (void)in_sizes; (void)n_in; (void)out_size;
    const int*   tokens  = (const int*)  d_in[0];
    const float* emb     = (const float*)d_in[1];
    const float* W_ih0   = (const float*)d_in[2];
    const float* W_hh0   = (const float*)d_in[3];
    const float* b_ih0   = (const float*)d_in[4];
    const float* b_hh0   = (const float*)d_in[5];
    const float* W_ih1   = (const float*)d_in[6];
    const float* W_hh1   = (const float*)d_in[7];
    const float* b_ih1   = (const float*)d_in[8];
    const float* b_hh1   = (const float*)d_in[9];
    const float* W_mean  = (const float*)d_in[10];
    const float* b_mean  = (const float*)d_in[11];
    const float* W_lstd  = (const float*)d_in[12];
    const float* b_lstd  = (const float*)d_in[13];

    // workspace layout (all offsets 256B aligned)
    char* ws = (char*)d_ws;
    const size_t OFF_BAR  = 0;                      // 256 B
    const size_t OFF_C0   = 256;                    // 64*1024 f32
    const size_t OFF_C1   = OFF_C0  + 262144;
    const size_t OFF_H0   = OFF_C1  + 262144;       // 2 x 64*1024 bf16 (ping-pong)
    const size_t OFF_H1   = OFF_H0  + 262144;
    const size_t ZERO_B   = OFF_H1  + 262144;       // zero everything above
    const size_t OFF_B0   = ZERO_B;                 // bias0 f32[4096]
    const size_t OFF_B1   = OFF_B0  + 16384;
    const size_t OFF_XBF  = OFF_B1  + 16384;        // 16384*512 bf16
    const size_t OFF_WIH0 = OFF_XBF + 16777216;     // 4096*512  bf16
    const size_t OFF_WHH0 = OFF_WIH0 + 4194304;     // 4096*1024 bf16
    const size_t OFF_WIH1 = OFF_WHH0 + 8388608;
    const size_t OFF_WHH1 = OFF_WIH1 + 8388608;
    const size_t OFF_WM   = OFF_WHH1 + 8388608;     // 128*4096 bf16
    const size_t OFF_WL   = OFF_WM  + 1048576;
    const size_t OFF_HID  = OFF_WL  + 1048576;      // 16384*4096 bf16
    const size_t TOTAL    = OFF_HID + 134217728;
    if (ws_size < TOTAL) return;

    unsigned* bar   = (unsigned*)(ws + OFF_BAR);
    float*    c0    = (float*)(ws + OFF_C0);
    float*    c1    = (float*)(ws + OFF_C1);
    bf16*     h0b   = (bf16*) (ws + OFF_H0);
    bf16*     h1b   = (bf16*) (ws + OFF_H1);
    float*    bias0 = (float*)(ws + OFF_B0);
    float*    bias1 = (float*)(ws + OFF_B1);
    bf16*     xbf   = (bf16*) (ws + OFF_XBF);
    bf16*     wih0  = (bf16*) (ws + OFF_WIH0);
    bf16*     whh0  = (bf16*) (ws + OFF_WHH0);
    bf16*     wih1  = (bf16*) (ws + OFF_WIH1);
    bf16*     whh1  = (bf16*) (ws + OFF_WHH1);
    bf16*     wm    = (bf16*) (ws + OFF_WM);
    bf16*     wl    = (bf16*) (ws + OFF_WL);
    bf16*     hid   = (bf16*) (ws + OFF_HID);

    // 1) zero barrier + c/h state
    {
        int n = (int)(ZERO_B / 4);
        zero_kernel<<<(n + 255) / 256, 256, 0, stream>>>((unsigned*)ws, n);
    }
    // 2) fold biases
    bias_sum_kernel<<<16, 256, 0, stream>>>(b_ih0, b_hh0, bias0, G4H);
    bias_sum_kernel<<<16, 256, 0, stream>>>(b_ih1, b_hh1, bias1, G4H);
    // 3) weights fp32 -> bf16 (L2-resident working set for the recurrence)
    cvt_bf16_kernel<<<(G4H * EE + 255) / 256, 256, 0, stream>>>(W_ih0, wih0, G4H * EE);
    cvt_bf16_kernel<<<(G4H * HH + 255) / 256, 256, 0, stream>>>(W_hh0, whh0, G4H * HH);
    cvt_bf16_kernel<<<(G4H * HH + 255) / 256, 256, 0, stream>>>(W_ih1, wih1, G4H * HH);
    cvt_bf16_kernel<<<(G4H * HH + 255) / 256, 256, 0, stream>>>(W_hh1, whh1, G4H * HH);
    cvt_bf16_kernel<<<(ZZ * HIDK + 255) / 256, 256, 0, stream>>>(W_mean, wm, ZZ * HIDK);
    cvt_bf16_kernel<<<(ZZ * HIDK + 255) / 256, 256, 0, stream>>>(W_lstd, wl, ZZ * HIDK);
    // 4) embedding gather + bf16 convert
    embed_kernel<<<(NROW * 64) / 256, 256, 0, stream>>>(tokens, emb, xbf);
    // 5) persistent LSTM recurrence (must be co-resident: 32 WGs)
    lstm_persistent<<<32, 256, 0, stream>>>(xbf, wih0, whh0, wih1, whh1,
                                            bias0, bias1, c0, c1, h0b, h1b, hid, bar);
    // 6) output projections -> d_out = [mean | log_std]
    proj_kernel<<<2048, 256, 0, stream>>>(hid, wm, wl, b_mean, b_lstd, (float*)d_out);
}
